// Codebook_12249246728357
// MI455X (gfx1250) — compile-verified, matrix operand order checked by reading, not run on hardware
//
#include <hip/hip_runtime.h>
#include <hip/hip_bf16.h>
#include <float.h>

typedef __attribute__((ext_vector_type(16))) __bf16 v16bf;
typedef __attribute__((ext_vector_type(8)))  float  v8f;
typedef int v4i __attribute__((vector_size(16)));   // matches builtin param type

// Problem dims (fixed by reference)
//  z: [16,256,64,64] f32, codebook: [1024,256] f32
//  N = 65536 rows, C = 256, K = 1024
//  out: zq NCHW (16777216 f32) ++ idx (65536 f32) ++ loss (1 f32)

// ---- CDNA5 async global->LDS copy (ASYNCcnt) with portable fallback --------
#if defined(__HIP_DEVICE_COMPILE__) && defined(__gfx1250__) && \
    __has_builtin(__builtin_amdgcn_global_load_async_to_lds_b128)
#define HAS_ASYNC_LDS 1
#else
#define HAS_ASYNC_LDS 0
#endif

__device__ __forceinline__ void copy16_g2lds(void* lds_dst, const void* gsrc)
{
#if HAS_ASYNC_LDS
    __builtin_amdgcn_global_load_async_to_lds_b128(
        (v4i*)gsrc, (v4i*)lds_dst, 0, 0);
#else
    *(float4*)lds_dst = *(const float4*)gsrc;
#endif
}

__device__ __forceinline__ void wait_async_lds()
{
#if HAS_ASYNC_LDS
#if __has_builtin(__builtin_amdgcn_s_wait_asynccnt)
    __builtin_amdgcn_s_wait_asynccnt(0);
#else
    asm volatile("s_wait_asynccnt 0x0" ::: "memory");
#endif
#endif
}

// ---------------------------------------------------------------------------
// Kernel 1: pre-swizzle codebook into WMMA B-fragment order (bf16) + norms.
//   lane = ((kk>>3)&1)*16 + n,  vgpr j = ((kk>>4)<<2) + ((kk&7)>>1), half = kk&1
// prep layout: [kb(64)][cb(8)][lane(32)][elem(16)] bf16  (512 KB total)
// ---------------------------------------------------------------------------
__global__ __launch_bounds__(256)
void vq_prep_kernel(const float* __restrict__ codebook,
                    __bf16* __restrict__ prep,
                    float* __restrict__ cbnorm,
                    float* __restrict__ loss_acc)
{
    __shared__ float sm[256];
    const int k = blockIdx.x;        // code 0..1023
    const int c = threadIdx.x;       // channel 0..255
    const float v = codebook[k * 256 + c];

    const int kb  = k >> 4, kn = k & 15;
    const int cbk = c >> 5, kk = c & 31;
    const int ln  = (((kk >> 3) & 1) << 4) + kn;
    const int jj  = ((kk >> 4) << 2) + ((kk & 7) >> 1);
    const int hh  = kk & 1;
    prep[(((size_t)(kb * 8 + cbk)) * 32 + ln) * 16 + jj * 2 + hh] = (__bf16)v;

    sm[c] = v * v;
    __syncthreads();
    for (int s = 128; s > 0; s >>= 1) {
        if (c < s) sm[c] += sm[c + s];
        __syncthreads();
    }
    if (c == 0) cbnorm[k] = sm[0];
    if (k == 0 && c == 0) *loss_acc = 0.0f;   // zero loss accumulator each call
}

// ---------------------------------------------------------------------------
// Kernel 2: argmin over codes via bf16 WMMA.
// Block = 128 rows (8 waves x one 16-row M-tile each), all 1024 codes.
// Codebook streamed once per block through double-buffered LDS via async
// global->LDS copies (2 kblocks = 16 KB per stage, one barrier per stage).
// B fragments software-pipelined by one cb step so each ds_load pair is
// covered by two WMMAs.  dist = ||cb||^2 - 2*dot (||zf||^2 constant per row).
// ---------------------------------------------------------------------------
__global__ __launch_bounds__(256)
void vq_argmin_kernel(const float* __restrict__ z,
                      const __bf16* __restrict__ prep,
                      const float* __restrict__ cbnorm,
                      int* __restrict__ idx_ws,
                      float* __restrict__ idx_out)
{
    __shared__ __align__(32) __bf16 tileA[8][8][32][16];   // 64 KB: 8 M-tiles
    __shared__ __align__(32) __bf16 bufB[2][8192];         // 2 x 16 KB (2 kblocks each)

    const int t    = threadIdx.x;
    const int wave = t >> 5;
    const int lane = t & 31;
    const int col  = lane & 15;         // code-within-kblock this lane owns

    const int n0  = blockIdx.x * 128;
    const int b   = n0 >> 12;           // /4096
    const int hw0 = n0 & 4095;
    const size_t zbase = (size_t)b * (256u * 4096u) + (size_t)hw0;

    // ---- Stage 128x256 z tile into LDS in WMMA A-fragment order (bf16) ----
    for (int it = t; it < 32768; it += 256) {
        const int c = it >> 7;          // channel 0..255
        const int m = it & 127;         // local row
        const float v = z[zbase + (size_t)c * 4096 + m];
        const int ti = m >> 4, mm = m & 15;
        const int cbk = c >> 5, kk = c & 31;
        const int ln = (((kk >> 3) & 1) << 4) + mm;
        const int jj = ((kk >> 4) << 2) + ((kk & 7) >> 1);
        const int hh = kk & 1;
        tileA[ti][cbk][ln][jj * 2 + hh] = (__bf16)v;
    }

    // ---- Kick off async stage of kblock-pair 0 into bufB[0] ----
    {
        const char* src = (const char*)(prep) + t * 64;       // 16 KB / 256 thr
        char*       dst = (char*)&bufB[0][0] + t * 64;
#pragma unroll
        for (int q = 0; q < 4; ++q)
            copy16_g2lds(dst + 16 * q, src + 16 * q);
    }

    __syncthreads();                     // tileA visible to all waves

    // A fragments for this wave's M-tile stay in registers for the whole loop
    v16bf afrag[8];
#pragma unroll
    for (int cb = 0; cb < 8; ++cb)
        afrag[cb] = *reinterpret_cast<const v16bf*>(&tileA[wave][cb][lane][0]);

    float bd[16];
    int   bi[16];
#pragma unroll
    for (int r = 0; r < 16; ++r) { bd[r] = FLT_MAX; bi[r] = 0; }

    for (int kp = 0; kp < 32; ++kp) {    // 32 kblock-pairs
        const int cur = kp & 1;
        wait_async_lds();                // my copies into bufB[cur] complete
        __syncthreads();                 // all copies done + prev compute done

        // issue next stage (async DMA overlaps the WMMA burst below)
        if (kp + 1 < 32) {
            const char* src = (const char*)(prep + (size_t)(kp + 1) * 8192) + t * 64;
            char*       dst = (char*)&bufB[1 - cur][0] + t * 64;
#pragma unroll
            for (int q = 0; q < 4; ++q)
                copy16_g2lds(dst + 16 * q, src + 16 * q);
        }

        // hoist the two per-kblock norm loads so they overlap the WMMAs
        const int   code0 = ((kp << 1) + 0) * 16 + col;
        const int   code1 = ((kp << 1) + 1) * 16 + col;
        const float cbn0  = cbnorm[code0];
        const float cbn1  = cbnorm[code1];

        const __bf16* bb = &bufB[cur][0];
        v8f acc0 = {};
        v8f acc1 = {};
        // software pipeline: load cb+1's fragments before issuing cb's WMMAs
        v16bf bf0 = *reinterpret_cast<const v16bf*>(bb + lane * 16);
        v16bf bf1 = *reinterpret_cast<const v16bf*>(bb + 4096 + lane * 16);
#pragma unroll
        for (int cb = 0; cb < 8; ++cb) {
            const v16bf c0 = bf0;
            const v16bf c1 = bf1;
            if (cb < 7) {
                bf0 = *reinterpret_cast<const v16bf*>(bb + (cb + 1) * 512 + lane * 16);
                bf1 = *reinterpret_cast<const v16bf*>(bb + 4096 + (cb + 1) * 512 + lane * 16);
            }
            acc0 = __builtin_amdgcn_wmma_f32_16x16x32_bf16(
                false, afrag[cb], false, c0, (short)0, acc0, false, false);
            acc1 = __builtin_amdgcn_wmma_f32_16x16x32_bf16(
                false, afrag[cb], false, c1, (short)0, acc1, false, false);
        }

#pragma unroll
        for (int r = 0; r < 8; ++r) {
            const float d0 = fmaf(-2.0f, acc0[r], cbn0);
            if (d0 < bd[r]) { bd[r] = d0; bi[r] = code0; }
            const float d1 = fmaf(-2.0f, acc1[r], cbn1);
            if (d1 < bd[8 + r]) { bd[8 + r] = d1; bi[8 + r] = code1; }
        }
    }

    // merge the two per-half candidates, then reduce across the 16 lanes of
    // each half (C-matrix: lane%16 = code column, lanes 0-15 rows r, 16-31 r+8)
#pragma unroll
    for (int r = 0; r < 8; ++r) {
        float d = bd[r]; int i = bi[r];
        if (bd[8 + r] < d || (bd[8 + r] == d && bi[8 + r] < i)) { d = bd[8 + r]; i = bi[8 + r]; }
#pragma unroll
        for (int m = 1; m <= 8; m <<= 1) {
            const float od = __shfl_xor(d, m, 32);
            const int   oi = __shfl_xor(i, m, 32);
            if (od < d || (od == d && oi < i)) { d = od; i = oi; }
        }
        bd[r] = d; bi[r] = i;
    }

    if (col == 0) {
        const int half = lane >> 4;      // lanes 0..15 -> rows r, 16..31 -> r+8
#pragma unroll
        for (int r = 0; r < 8; ++r) {
            const int n = n0 + wave * 16 + r + (half << 3);
            idx_ws[n]  = bi[r];
            idx_out[n] = (float)bi[r];
        }
    }
}

// ---------------------------------------------------------------------------
// Kernel 3: gather codebook[idx] -> NCHW output, fused loss partial sums.
// block = 32 hw positions x 8 channels; writes are 128B-coalesced per channel.
// ---------------------------------------------------------------------------
__global__ __launch_bounds__(256)
void vq_gather_kernel(const float* __restrict__ z,
                      const float* __restrict__ codebook,
                      const int* __restrict__ idx_ws,
                      float* __restrict__ out_zq,
                      float* __restrict__ loss_acc)
{
    __shared__ float sm[256];
    const int g   = blockIdx.x;          // 0..2047: (b, hw chunk)
    const int b   = g >> 7;              // /128
    const int hw0 = (g & 127) << 5;      // *32
    const int t   = threadIdx.x;
    const int hwl = t & 31;
    const int cl  = t >> 5;              // 0..7
    const int c   = (blockIdx.y << 3) + cl;

    const int n = (b << 12) + hw0 + hwl;
    const int k = idx_ws[n];

    const float v  = codebook[k * 256 + c];
    const size_t zi = ((size_t)(b * 256 + c) << 12) + (size_t)(hw0 + hwl);
    out_zq[zi] = v;

    const float diff = v - z[zi];
    sm[t] = diff * diff;
    __syncthreads();
    for (int s = 128; s > 0; s >>= 1) {
        if (t < s) sm[t] += sm[t + s];
        __syncthreads();
    }
    if (t == 0) atomicAdd(loss_acc, sm[0]);
}

// ---------------------------------------------------------------------------
// Kernel 4: finalize scalar loss = 0.75 * mean(diff^2)
// ---------------------------------------------------------------------------
__global__ void vq_loss_kernel(const float* __restrict__ loss_acc,
                               float* __restrict__ out_loss)
{
    *out_loss = 0.75f * (*loss_acc) / 16777216.0f;
}

// ---------------------------------------------------------------------------
extern "C" void kernel_launch(void* const* d_in, const int* in_sizes, int n_in,
                              void* d_out, int out_size, void* d_ws, size_t ws_size,
                              hipStream_t stream)
{
    const float* z        = (const float*)d_in[0];   // [16,256,64,64]
    const float* codebook = (const float*)d_in[1];   // [1024,256]

    float* out      = (float*)d_out;
    float* out_idx  = out + 16777216;                // 65536 floats
    float* out_loss = out + 16777216 + 65536;        // 1 float

    char*   ws       = (char*)d_ws;
    __bf16* prep     = (__bf16*)ws;                         // 524288 B
    float*  cbnorm   = (float*)(ws + 524288);               //   4096 B
    int*    idx_ws   = (int*)  (ws + 524288 + 4096);        // 262144 B
    float*  loss_acc = (float*)(ws + 524288 + 4096 + 262144);

    vq_prep_kernel<<<1024, 256, 0, stream>>>(codebook, prep, cbnorm, loss_acc);
    vq_argmin_kernel<<<512, 256, 0, stream>>>(z, prep, cbnorm, idx_ws, out_idx);
    vq_gather_kernel<<<dim3(2048, 32), 256, 0, stream>>>(z, codebook, idx_ws, out, loss_acc);
    vq_loss_kernel<<<1, 1, 0, stream>>>(loss_acc, out_loss);
}
